// MatchingHead_78288663872403
// MI455X (gfx1250) — compile-verified, matrix operand order checked by reading, not run on hardware
//
#include <hip/hip_runtime.h>
#include <math.h>

typedef _Float16 half8 __attribute__((ext_vector_type(8)));
typedef _Float16 v16h  __attribute__((ext_vector_type(16)));
typedef float    v8f   __attribute__((ext_vector_type(8)));
typedef float    v4f   __attribute__((ext_vector_type(4)));

#define BATCH   4
#define NROW    4096
#define DIM     256
#define TOPK    32
#define CHUNK   256                 // columns per chunk (8 waves x 2 tiles x 16)
#define NCHUNKS (NROW / CHUNK)      // 16
#define SSTRIDE 260                 // padded score-row stride (breaks bank aliasing)

#if __has_builtin(__builtin_amdgcn_sched_group_barrier)
#define SGB(mask, size, id) __builtin_amdgcn_sched_group_barrier(mask, size, id)
#else
#define SGB(mask, size, id)
#endif
// sched_group_barrier categories: 0x008 = MFMA/WMMA, 0x020 = VMEM read

// Build a 16x32 f16 A/B fragment for one K-chunk.
// Lane layout (ISA 7.12.2, 16-bit A 16x32): lanes 0-15 row M=lane, K=0..7,16..23;
// lanes 16-31 same rows, K=8..15,24..31.  Two 16-byte row loads per fragment.
__device__ __forceinline__ v16h load_frag(const _Float16* __restrict__ rowPtr,
                                          int kc, int hi) {
    half8 lo = *(const half8*)(rowPtr + kc * 32 + hi * 8);
    half8 hs = *(const half8*)(rowPtr + kc * 32 + 16 + hi * 8);
    return __builtin_shufflevector(lo, hs, 0, 1, 2, 3, 4, 5, 6, 7,
                                           8, 9, 10, 11, 12, 13, 14, 15);
}

// One wave per row: L2-normalize fp32 row of 256, emit f16.
__global__ void normalize_kernel(const float* __restrict__ F,
                                 _Float16* __restrict__ out) {
    const int row  = blockIdx.x;
    const int lane = threadIdx.x;           // 0..31, 8 elements each
    const float* src = F + (size_t)row * DIM + lane * 8;
    float4 a = *(const float4*)(src);
    float4 b = *(const float4*)(src + 4);
    float ss = a.x * a.x + a.y * a.y + a.z * a.z + a.w * a.w +
               b.x * b.x + b.y * b.y + b.z * b.z + b.w * b.w;
    for (int off = 16; off > 0; off >>= 1) ss += __shfl_xor(ss, off, 32);
    const float scale = 1.0f / fmaxf(sqrtf(ss), 1e-12f);
    half8 h;
    h[0] = (_Float16)(a.x * scale); h[1] = (_Float16)(a.y * scale);
    h[2] = (_Float16)(a.z * scale); h[3] = (_Float16)(a.w * scale);
    h[4] = (_Float16)(b.x * scale); h[5] = (_Float16)(b.y * scale);
    h[6] = (_Float16)(b.z * scale); h[7] = (_Float16)(b.w * scale);
    *(half8*)(out + (size_t)row * DIM + lane * 8) = h;
}

// Stream 268 MB of zeros with non-temporal hints: don't evict the f16
// operands / L2-resident GEMM working set.
__global__ void zero_kernel(v4f* __restrict__ out) {
    size_t i = (size_t)blockIdx.x * blockDim.x + threadIdx.x;
    v4f z = {0.f, 0.f, 0.f, 0.f};
    __builtin_nontemporal_store(z, out + i);
}

// One block = one (batch, 16-row tile).  8 waves; wave w computes column tiles
// {w, w+8} of each 256-wide chunk, then rows {w, w+8} of the per-row top-32.
__launch_bounds__(256, 1)
__global__ void gemm_topk_kernel(const _Float16* __restrict__ f1h,
                                 const _Float16* __restrict__ f2h,
                                 float* __restrict__ soft_corr,
                                 int* __restrict__ topk_idx) {
    __shared__ float scoreS[16 * SSTRIDE];
    __shared__ float tkS[16 * TOPK];
    __shared__ int   tkI[16 * TOPK];

    const int tid  = threadIdx.x;
    const int wave = tid >> 5;
    const int lane = tid & 31;
    const int m    = lane & 15;
    const int hi   = lane >> 4;

    const int batch   = blockIdx.x / (NROW / 16);
    const int mtile   = blockIdx.x % (NROW / 16);
    const int rowBase = mtile * 16;

    for (int i = tid; i < 16 * TOPK; i += 256) { tkS[i] = -INFINITY; tkI[i] = 0; }

    // A fragments: 16 rows x 256 f16, held in VGPRs across every chunk.
    const _Float16* aRow = f1h + ((size_t)(batch * NROW + rowBase + m)) * DIM;
    v16h aFrag[8];
#pragma unroll
    for (int kc = 0; kc < 8; ++kc) aFrag[kc] = load_frag(aRow, kc, hi);

    __syncthreads();

    const float invT = 1.0f / 0.07f;

    for (int ch = 0; ch < NCHUNKS; ++ch) {
        const int colBase = ch * CHUNK + wave * 16;
        const _Float16* bRow0 =
            f2h + ((size_t)(batch * NROW + colBase + m)) * DIM;
        const _Float16* bRow1 = bRow0 + (size_t)128 * DIM;

        // Kick next chunk's cachelines toward L2 before this chunk's math.
        if (ch + 1 < NCHUNKS) {
            __builtin_prefetch(bRow0 + (size_t)CHUNK * DIM, 0, 1);
            __builtin_prefetch(bRow1 + (size_t)CHUNK * DIM, 0, 1);
        }

        v8f acc0 = {};
        v8f acc1 = {};

        // All 32 B loads + 16 WMMAs in one scheduling region; the
        // sched_group_barrier recipe below interleaves them at distance 4
        // fragments so each WMMA waits with loadcnt<=6 instead of 0.
        v16h b0f[8], b1f[8];
#pragma unroll
        for (int kc = 0; kc < 8; ++kc) {
            b0f[kc] = load_frag(bRow0, kc, hi);
            b1f[kc] = load_frag(bRow1, kc, hi);
        }
#pragma unroll
        for (int kc = 0; kc < 8; ++kc) {
            acc0 = __builtin_amdgcn_wmma_f32_16x16x32_f16(
                false, aFrag[kc], false, b0f[kc], (short)0, acc0, false, false);
            acc1 = __builtin_amdgcn_wmma_f32_16x16x32_f16(
                false, aFrag[kc], false, b1f[kc], (short)0, acc1, false, false);
        }
        // Prescribed schedule: 8 loads | 12 x {1 wmma, 2 loads} | 4 wmmas.
        SGB(0x020, 8, 0);
#pragma unroll
        for (int g = 0; g < 12; ++g) {
            SGB(0x008, 1, 0);
            SGB(0x020, 2, 0);
        }
        SGB(0x008, 4, 0);

        // C layout: lanes 0-15 -> N=lane, M=vgpr; lanes 16-31 -> N=lane-16, M=8+vgpr
#pragma unroll
        for (int i = 0; i < 8; ++i) {
            const int r = hi * 8 + i;
            scoreS[r * SSTRIDE + wave * 16 + m]       = acc0[i] * invT;
            scoreS[r * SSTRIDE + 128 + wave * 16 + m] = acc1[i] * invT;
        }
        __syncthreads();

        // Top-k maintenance: wave owns rows {wave, wave+8}.
#pragma unroll
        for (int rep = 0; rep < 2; ++rep) {
            const int r = wave + rep * 8;
            float cs[8]; int cc[8];
#pragma unroll
            for (int j = 0; j < 8; ++j) {
                const int c = lane + j * 32;
                cs[j] = scoreS[r * SSTRIDE + c];
                cc[j] = ch * CHUNK + c;
            }
            while (true) {
                float best = -INFINITY; int bcol = 0x7FFFFFFF; int bslot = 0;
#pragma unroll
                for (int j = 0; j < 8; ++j)
                    if (cs[j] > best) { best = cs[j]; bcol = cc[j]; bslot = j; }
                int bown = (lane << 3) | bslot;
                for (int off = 16; off > 0; off >>= 1) {
                    float os = __shfl_xor(best, off, 32);
                    int   oc = __shfl_xor(bcol, off, 32);
                    int   oo = __shfl_xor(bown, off, 32);
                    if (os > best || (os == best && oc < bcol)) {
                        best = os; bcol = oc; bown = oo;
                    }
                }
                if (!(best > tkS[r * TOPK + (TOPK - 1)])) break;  // uniform
                // Wave-parallel rank insert into sorted (descending) list.
                float sl = tkS[r * TOPK + lane];
                int   il = tkI[r * TOPK + lane];
                int pos = (int)__popcll(
                    __ballot(sl > best || (sl == best && il < bcol)));
                if (lane >= pos && lane < TOPK - 1) {
                    tkS[r * TOPK + lane + 1] = sl;
                    tkI[r * TOPK + lane + 1] = il;
                }
                if (lane == pos) {
                    tkS[r * TOPK + pos] = best;
                    tkI[r * TOPK + pos] = bcol;
                }
                if ((bown >> 3) == lane) cs[bown & 7] = -INFINITY;
            }
        }
        __syncthreads();
    }

    // Softmax over the sorted top-32 (max = slot 0) + scatter.
#pragma unroll
    for (int rep = 0; rep < 2; ++rep) {
        const int r    = wave + rep * 8;
        const int grow = batch * NROW + rowBase + r;
        float s  = tkS[r * TOPK + lane];
        int  idx = tkI[r * TOPK + lane];
        float mx = tkS[r * TOPK];
        float e  = __expf(s - mx);
        float sum = e;
        for (int off = 16; off > 0; off >>= 1) sum += __shfl_xor(sum, off, 32);
        __builtin_nontemporal_store(e / sum,
                                    soft_corr + (size_t)grow * NROW + idx);
        topk_idx[grow * TOPK + lane] = idx;
    }
}

extern "C" void kernel_launch(void* const* d_in, const int* in_sizes, int n_in,
                              void* d_out, int out_size, void* d_ws, size_t ws_size,
                              hipStream_t stream) {
    const float* F1 = (const float*)d_in[0];
    const float* F2 = (const float*)d_in[1];

    float* soft_corr = (float*)d_out;
    int*   topk_idx  = (int*)((float*)d_out + (size_t)BATCH * NROW * NROW);

    _Float16* f1h = (_Float16*)d_ws;
    _Float16* f2h = f1h + (size_t)BATCH * NROW * DIM;

    normalize_kernel<<<BATCH * NROW, 32, 0, stream>>>(F1, f1h);
    normalize_kernel<<<BATCH * NROW, 32, 0, stream>>>(F2, f2h);

    const size_t n4 = (size_t)BATCH * NROW * NROW / 4;   // 16,777,216 float4s
    zero_kernel<<<(unsigned)(n4 / 256), 256, 0, stream>>>((v4f*)d_out);

    gemm_topk_kernel<<<BATCH * (NROW / 16), 256, 0, stream>>>(
        f1h, f2h, soft_corr, topk_idx);
}
